// HiPLayer_46600395161752
// MI455X (gfx1250) — matrix operand order repaired and checked by reading, not compiled, not run on hardware
//
#include <hip/hip_runtime.h>

// ---------------------------------------------------------------------------
// CDNA5 (gfx1250): grouped projection -> latent cross-attention -> 2x
// transformer layers.  All GEMMs on v_wmma_f32_16x16x32_bf16.  wave32,
// 256-thread (8-wave) blocks.  TDM (tensor_load_to_lds) stages the x4 tile
// in the latent kernel.  f32->bf16 via vector fptrunc (v_cvt_pk_bf16_f32).
// ---------------------------------------------------------------------------

typedef __attribute__((ext_vector_type(16))) __bf16 v16bf;
typedef __attribute__((ext_vector_type(2)))  __bf16 v2bf;
typedef __attribute__((ext_vector_type(8)))  float  v8f;
typedef __attribute__((ext_vector_type(2)))  float  v2f;
typedef __attribute__((ext_vector_type(4)))  unsigned int v4u;
typedef __attribute__((ext_vector_type(8)))  int    v8i;
typedef __attribute__((ext_vector_type(4)))  int    v4i;

#ifndef USE_TDM
#define USE_TDM 1
#endif
#if USE_TDM && defined(__has_builtin)
#if __has_builtin(__builtin_amdgcn_tensor_load_to_lds) && \
    __has_builtin(__builtin_amdgcn_s_wait_tensorcnt)
#define TDM_OK 1
#endif
#endif

#define LANE (threadIdx.x & 31)
#define WAVE (threadIdx.x >> 5)

// ---- helpers --------------------------------------------------------------
__device__ __forceinline__ unsigned short f2bf(float f) {
  return __builtin_bit_cast(unsigned short, (__bf16)f);
}

// Packed pair conversion via 2-wide vector fptrunc -> one v_cvt_pk_bf16_f32.
__device__ __forceinline__ unsigned int f2bf_pk(float lo, float hi) {
  v2f in = {lo, hi};
  v2bf out = __builtin_convertvector(in, v2bf);
  return __builtin_bit_cast(unsigned int, out);
}

__device__ __forceinline__ uint2 f2bf_pk4(float4 f) {
  return make_uint2(f2bf_pk(f.x, f.y), f2bf_pk(f.z, f.w));
}

__device__ __forceinline__ v8f v8f_zero() {
  v8f z;
#pragma unroll
  for (int i = 0; i < 8; ++i) z[i] = 0.0f;
  return z;
}

// ---- WMMA fragment loaders (CDNA5 ISA §7.12.2) -----------------
__device__ __forceinline__ v16bf load_a_frag(const unsigned short* lds, int ld,
                                             int mBase, int kBase) {
  const int lane = LANE;
  const int m  = mBase + (lane & 15);
  const int hi = lane >> 4;
  v16bf a;
#pragma unroll
  for (int e = 0; e < 16; ++e) {
    const int v = e >> 1;
    const int k = hi * 16 + (v << 1) + (e & 1);
    a[e] = __builtin_bit_cast(__bf16, lds[m * ld + kBase + k]);
  }
  return a;
}

__device__ __forceinline__ v16bf load_b_frag_nmajor(const unsigned short* lds, int ld,
                                                    int nBase, int kBase) {
  const int lane = LANE;
  const int n  = nBase + (lane & 15);
  const int hi = lane >> 4;
  v16bf b;
#pragma unroll
  for (int e = 0; e < 16; ++e)
    b[e] = __builtin_bit_cast(__bf16, lds[n * ld + kBase + hi * 16 + e]);
  return b;
}

__device__ __forceinline__ v16bf load_b_frag_kmajor(const unsigned short* lds, int ld,
                                                    int kBase, int nBase) {
  const int lane = LANE;
  const int n  = nBase + (lane & 15);
  const int hi = lane >> 4;
  v16bf b;
#pragma unroll
  for (int e = 0; e < 16; ++e)
    b[e] = __builtin_bit_cast(__bf16, lds[(kBase + hi * 16 + e) * ld + n]);
  return b;
}

__device__ __forceinline__ v8f wmma_bf16(v16bf a, v16bf b, v8f c) {
  return __builtin_amdgcn_wmma_f32_16x16x32_bf16(false, a, false, b,
                                                 (short)0, c, false, false);
}

// ---------------------------------------------------------------------------
// Kernel 1: X4(bf16) = X(fp32, MxKd) @ W(fp32, KdxN, k-major) + bias
// Register-pipelined staging: tile k+1 global loads overlap tile k WMMAs.
// ---------------------------------------------------------------------------
__global__ __launch_bounds__(256) void k_proj(const float* __restrict__ X,
                                              const float* __restrict__ W,
                                              const float* __restrict__ bias,
                                              unsigned short* __restrict__ X4,
                                              int M, int Kd, int N) {
  (void)M;
  __shared__ unsigned short As[128 * 40];   // [m][k]
  __shared__ unsigned short Bt[128 * 40];   // [n][k] (transposed W tile)
  const int tid = threadIdx.x;
  const int mB = blockIdx.x * 128;
  const int nB = blockIdx.y * 128;
  const int w = WAVE, lane = LANE;
  const int wr = (w >> 1) * 32;
  const int wc = (w & 1) * 64;

  // per-thread staging coordinates (loop-invariant)
  const int ar = tid >> 3, ac4 = (tid & 7) << 2;          // A: float4 per thread
  const int wn = tid & 127, wkh = (tid >> 7) * 16;        // W: 16 contiguous k
  const float* Arow = X + (size_t)(mB + ar) * Kd + ac4;

  float4 ra[4];
  float  rb[16];
  auto load_tiles = [&](int k0) {
#pragma unroll
    for (int t = 0; t < 4; ++t)
      ra[t] = *(const float4*)&Arow[(size_t)(t * 32) * Kd + k0];
#pragma unroll
    for (int t = 0; t < 16; ++t)
      rb[t] = W[(size_t)(k0 + wkh + t) * N + nB + wn];
  };
  auto store_tiles = [&]() {
#pragma unroll
    for (int t = 0; t < 4; ++t)
      *(uint2*)&As[(ar + t * 32) * 40 + ac4] = f2bf_pk4(ra[t]);
#pragma unroll
    for (int t = 0; t < 8; ++t)
      *(unsigned int*)&Bt[wn * 40 + wkh + 2 * t] = f2bf_pk(rb[2 * t], rb[2 * t + 1]);
  };

  v8f acc[2][4];
#pragma unroll
  for (int i = 0; i < 2; ++i)
#pragma unroll
    for (int j = 0; j < 4; ++j) acc[i][j] = v8f_zero();

  load_tiles(0);
  for (int k0 = 0; k0 < Kd; k0 += 32) {
    store_tiles();
    __syncthreads();
    if (k0 + 32 < Kd) {
      __builtin_prefetch(&Arow[(size_t)96 * Kd + k0 + 32], 0, 0);
      load_tiles(k0 + 32);                    // overlap with WMMA below
    }
#pragma unroll
    for (int i = 0; i < 2; ++i) {
      v16bf a = load_a_frag(As, 40, wr + i * 16, 0);
#pragma unroll
      for (int j = 0; j < 4; ++j) {
        v16bf b = load_b_frag_nmajor(Bt, 40, wc + j * 16, 0);
        acc[i][j] = wmma_bf16(a, b, acc[i][j]);
      }
    }
    __syncthreads();
  }
#pragma unroll
  for (int i = 0; i < 2; ++i)
#pragma unroll
    for (int j = 0; j < 4; ++j) {
      const int n = nB + wc + j * 16 + (lane & 15);
      const int m0 = mB + wr + i * 16 + (lane >> 4) * 8;
      const float bv = bias[n];
#pragma unroll
      for (int r = 0; r < 8; ++r)
        X4[(size_t)(m0 + r) * N + n] = f2bf(acc[i][j][r] + bv);
    }
}

// ---------------------------------------------------------------------------
// Kernel 2: latent cross-attention, one workgroup per (b,g).
// scores = emb[g] @ x4^T / 16; softmax over the 128 latents; y = attn @ x4.
// x4 chunk staged to LDS by the Tensor Data Mover.
// ---------------------------------------------------------------------------
__global__ __launch_bounds__(256) void k_latent(const unsigned short* __restrict__ X4,
                                                const float* __restrict__ emb,
                                                float* __restrict__ Y,
                                                float* __restrict__ AttnOut) {
  extern __shared__ char smem[];
  unsigned short* emb_s  = (unsigned short*)smem;        // 128 x 264
  unsigned short* x4c    = emb_s + 128 * 264;            // 64  x 264 (pad 16B/row)
  unsigned short* attn_s = x4c + 64 * 264;               // 128 x 72
  float*          sc     = (float*)(attn_s + 128 * 72);  // 128 x 68
  float*          red    = sc + 128 * 68;                // 256
  float*          inv_s  = red + 256;                    // 64

  const int tid = threadIdx.x, w = WAVE, lane = LANE;
  const int bg = blockIdx.x;
  const int g  = bg & 15;
  const unsigned short* X4g = X4 + (size_t)bg * 1024 * 256;

  {  // stage emb[g] as bf16 (float4 loads, packed-pair bf16 stores)
    const int r = tid >> 1, c4 = (tid & 1) << 7;  // 2 threads/row, 128 cols each
#pragma unroll
    for (int t = 0; t < 32; ++t) {
      float4 f = *(const float4*)&emb[((size_t)g * 128 + r) * 256 + c4 + t * 4];
      *(uint2*)&emb_s[r * 264 + c4 + t * 4] = f2bf_pk4(f);
    }
  }

  v8f yacc[16];
#pragma unroll
  for (int j = 0; j < 16; ++j) yacc[j] = v8f_zero();
  const float rscale = 0.0625f;  // 1/sqrt(256)

  for (int tB = 0; tB < 1024; tB += 64) {
    __syncthreads();  // emb staged (first iter); x4c/attn_s consumers done
#if defined(TDM_OK)
    if (tid < 32) {
      // --- Tensor DMA: 64 x 512B tile, +16B LDS pad per row (stride 264 u16)
      const unsigned lds_addr = (unsigned)(size_t)(void*)x4c;
      const unsigned long long ga =
          (unsigned long long)(size_t)(const void*)(X4g + (size_t)tB * 256);
      v4u g0;
      g0[0] = 1u;                                        // count=1
      g0[1] = lds_addr;                                  // lds_addr
      g0[2] = (unsigned)(ga & 0xFFFFFFFFu);              // global_addr lo
      g0[3] = (unsigned)((ga >> 32) & 0x1FFFFFFu) | (2u << 30);  // hi | type=2
      v8i g1;
      g1[0] = (int)((1u << 16) |       // data_size = 1 (2B)
                    (1u << 20) |       // pad_enable
                    (6u << 22) |       // pad_interval: 128 dwords (512B)
                    (3u << 25));       // pad_amount:   4 dwords (16B)
      g1[1] = (int)(256u << 16);       // tensor_dim0 = 256 (lo16 @ bits63:48)
      g1[2] = (int)(64u << 16);        // dim0 hi16=0 | tensor_dim1 = 64
      g1[3] = (int)(256u << 16);       // dim1 hi16=0 | tile_dim0 = 256
      g1[4] = 64;                      // tile_dim1 = 64 | tile_dim2 = 0
      g1[5] = 256;                     // tensor_dim0_stride lo32
      g1[6] = 0;                       // dim0_stride hi16 | dim1_stride lo16
      g1[7] = 0;
      v4i gz4 = {0, 0, 0, 0};
      v8i gz8 = {0, 0, 0, 0, 0, 0, 0, 0};
      __builtin_amdgcn_tensor_load_to_lds(g0, g1, gz4, gz4, gz8, 0);
      __builtin_amdgcn_s_wait_tensorcnt(0);
    }
#else
    {  // manual staging fallback: 8B copies, fully coalesced
      const int t = tid >> 2, c4 = (tid & 3) << 6;
#pragma unroll
      for (int q = 0; q < 16; ++q)
        *(ushort4*)&x4c[t * 264 + c4 + q * 4] =
            *(const ushort4*)&X4g[(size_t)(tB + t) * 256 + c4 + q * 4];
    }
#endif
    __syncthreads();

    // scores tile: wave w -> latent rows [16w,16w+16), cols [0,64)
    v8f sacc[4];
#pragma unroll
    for (int j = 0; j < 4; ++j) sacc[j] = v8f_zero();
    for (int k0 = 0; k0 < 256; k0 += 32) {
      v16bf a = load_a_frag(emb_s, 264, w * 16, k0);
#pragma unroll
      for (int j = 0; j < 4; ++j) {
        v16bf b = load_b_frag_nmajor(x4c, 264, j * 16, k0);  // n=t, k=c
        sacc[j] = wmma_bf16(a, b, sacc[j]);
      }
    }
#pragma unroll
    for (int j = 0; j < 4; ++j) {
      const int n  = j * 16 + (lane & 15);
      const int m0 = w * 16 + (lane >> 4) * 8;
#pragma unroll
      for (int r = 0; r < 8; ++r) sc[(m0 + r) * 68 + n] = sacc[j][r] * rscale;
    }
    __syncthreads();

    // softmax over the LATENT axis (128 rows), 4 partial segments per column
    {
      const int col = tid & 63, seg = tid >> 6;
      float mx = -3.4e38f;
      for (int r = seg * 32; r < seg * 32 + 32; ++r)
        mx = fmaxf(mx, sc[r * 68 + col]);
      red[tid] = mx;
    }
    __syncthreads();
    if (tid < 64)
      inv_s[tid] = fmaxf(fmaxf(red[tid], red[tid + 64]),
                         fmaxf(red[tid + 128], red[tid + 192]));
    __syncthreads();
    {
      const int col = tid & 63, seg = tid >> 6;
      const float mx = inv_s[col];
      float sum = 0.0f;
      for (int r = seg * 32; r < seg * 32 + 32; ++r) {
        const float e = __expf(sc[r * 68 + col] - mx);
        sc[r * 68 + col] = e;
        sum += e;
      }
      red[tid] = sum;
    }
    __syncthreads();
    if (tid < 64)
      inv_s[tid] = 1.0f / (red[tid] + red[tid + 64] + red[tid + 128] + red[tid + 192]);
    __syncthreads();
    // normalize + bf16 for WMMA + coalesced fp32 attn output
    for (int i = tid; i < 128 * 64; i += 256) {
      const int r = i >> 6, t = i & 63;
      const float av = sc[r * 68 + t] * inv_s[t];
      attn_s[r * 72 + t] = f2bf(av);
      AttnOut[((size_t)bg * 128 + r) * 1024 + tB + t] = av;
    }
    __syncthreads();

    // y(128x256) += attn(128x64) @ x4c(64x256)
#pragma unroll
    for (int k0 = 0; k0 < 64; k0 += 32) {
      v16bf a = load_a_frag(attn_s, 72, w * 16, k0);
#pragma unroll
      for (int j = 0; j < 16; ++j) {
        v16bf b = load_b_frag_kmajor(x4c, 264, k0, j * 16);  // k=t, n=c
        yacc[j] = wmma_bf16(a, b, yacc[j]);
      }
    }
  }

#pragma unroll
  for (int j = 0; j < 16; ++j) {
    const int n  = j * 16 + (lane & 15);
    const int m0 = w * 16 + (lane >> 4) * 8;
#pragma unroll
    for (int r = 0; r < 8; ++r)
      Y[((size_t)bg * 128 + m0 + r) * 256 + n] = yacc[j][r];
  }
}

// ---------------------------------------------------------------------------
// Kernel 3: Out(fp32, MxN) = A(fp32, MxKd) @ W^T + bias, W is (N x Kd) row-major.
// Both A and W rows are k-contiguous -> float4 staging, register-pipelined.
// ---------------------------------------------------------------------------
__global__ __launch_bounds__(256) void k_gemm_nt(const float* __restrict__ A,
                                                 const float* __restrict__ W,
                                                 const float* __restrict__ bias,
                                                 float* __restrict__ Out,
                                                 int M, int Kd, int N) {
  (void)M;
  __shared__ unsigned short As[128 * 40];
  __shared__ unsigned short Bt[128 * 40];   // [n][k]
  const int tid = threadIdx.x;
  const int mB = blockIdx.x * 128;
  const int nB = blockIdx.y * 128;
  const int w = WAVE, lane = LANE;
  const int wr = (w >> 1) * 32;
  const int wc = (w & 1) * 64;

  const int ar = tid >> 3, ac4 = (tid & 7) << 2;
  const float* Arow = A + (size_t)(mB + ar) * Kd + ac4;
  const float* Wrow = W + (size_t)(nB + ar) * Kd + ac4;

  float4 ra[4], rb[4];
  auto load_tiles = [&](int k0) {
#pragma unroll
    for (int t = 0; t < 4; ++t) {
      ra[t] = *(const float4*)&Arow[(size_t)(t * 32) * Kd + k0];
      rb[t] = *(const float4*)&Wrow[(size_t)(t * 32) * Kd + k0];
    }
  };
  auto store_tiles = [&]() {
#pragma unroll
    for (int t = 0; t < 4; ++t) {
      *(uint2*)&As[(ar + t * 32) * 40 + ac4] = f2bf_pk4(ra[t]);
      *(uint2*)&Bt[(ar + t * 32) * 40 + ac4] = f2bf_pk4(rb[t]);
    }
  };

  v8f acc[2][4];
#pragma unroll
  for (int i = 0; i < 2; ++i)
#pragma unroll
    for (int j = 0; j < 4; ++j) acc[i][j] = v8f_zero();

  load_tiles(0);
  for (int k0 = 0; k0 < Kd; k0 += 32) {
    store_tiles();
    __syncthreads();
    if (k0 + 32 < Kd) load_tiles(k0 + 32);    // overlap with WMMA below
#pragma unroll
    for (int i = 0; i < 2; ++i) {
      v16bf a = load_a_frag(As, 40, wr + i * 16, 0);
#pragma unroll
      for (int j = 0; j < 4; ++j) {
        v16bf b = load_b_frag_nmajor(Bt, 40, wc + j * 16, 0);
        acc[i][j] = wmma_bf16(a, b, acc[i][j]);
      }
    }
    __syncthreads();
  }
#pragma unroll
  for (int i = 0; i < 2; ++i)
#pragma unroll
    for (int j = 0; j < 4; ++j) {
      const int n = nB + wc + j * 16 + (lane & 15);
      const int m0 = mB + wr + i * 16 + (lane >> 4) * 8;
      const float bv = bias[n];
#pragma unroll
      for (int r = 0; r < 8; ++r)
        Out[(size_t)(m0 + r) * N + n] = acc[i][j][r] + bv;
    }
}

// ---------------------------------------------------------------------------
// Kernel 4: MHA core, one workgroup per (batch nb, head h).  S=128, Dh=32.
// ---------------------------------------------------------------------------
__global__ __launch_bounds__(256) void k_mha(const float* __restrict__ QKV,
                                             float* __restrict__ O) {
  extern __shared__ char smem[];
  unsigned short* qs = (unsigned short*)smem;     // 128 x 40
  unsigned short* ks = qs + 128 * 40;             // 128 x 40
  unsigned short* vs = ks + 128 * 40;             // 128 x 40 ([t][d], k-major)
  unsigned short* at = vs + 128 * 40;             // 128 x 136
  float*          s2 = (float*)(at + 128 * 136);  // 128 x 132

  const int tid = threadIdx.x, w = WAVE, lane = LANE;
  const int nb = blockIdx.x >> 3;
  const int h  = blockIdx.x & 7;

  {  // float4 staging of q,k,v slices (packed-pair bf16 stores)
    const int s = tid >> 1, d4 = (tid & 1) << 4;  // 2 threads/row, 16 d each
#pragma unroll
    for (int t = 0; t < 4; ++t) {
      const size_t base = ((size_t)s * 128 + nb) * 768 + h * 32 + d4 + t * 4;
      *(uint2*)&qs[s * 40 + d4 + t * 4] = f2bf_pk4(*(const float4*)&QKV[base]);
      *(uint2*)&ks[s * 40 + d4 + t * 4] = f2bf_pk4(*(const float4*)&QKV[base + 256]);
      *(uint2*)&vs[s * 40 + d4 + t * 4] = f2bf_pk4(*(const float4*)&QKV[base + 512]);
    }
  }
  __syncthreads();

  const float scale = 0.17677669529663687f;  // 1/sqrt(32)
  v8f sacc[8];
#pragma unroll
  for (int j = 0; j < 8; ++j) sacc[j] = v8f_zero();
  {
    v16bf a = load_a_frag(qs, 40, w * 16, 0);  // single K=32 step (Dh=32)
#pragma unroll
    for (int j = 0; j < 8; ++j) {
      v16bf b = load_b_frag_nmajor(ks, 40, j * 16, 0);  // n=t, k=d
      sacc[j] = wmma_bf16(a, b, sacc[j]);
    }
  }
#pragma unroll
  for (int j = 0; j < 8; ++j) {
    const int n  = j * 16 + (lane & 15);
    const int m0 = w * 16 + (lane >> 4) * 8;
#pragma unroll
    for (int r = 0; r < 8; ++r) s2[(m0 + r) * 132 + n] = sacc[j][r] * scale;
  }
  __syncthreads();

  if (tid < 128) {  // row softmax over the last axis
    float mx = -3.4e38f;
    for (int t = 0; t < 128; ++t) mx = fmaxf(mx, s2[tid * 132 + t]);
    float sum = 0.0f;
    for (int t = 0; t < 128; ++t) {
      const float e = __expf(s2[tid * 132 + t] - mx);
      s2[tid * 132 + t] = e;
      sum += e;
    }
    const float inv = 1.0f / sum;
#pragma unroll 4
    for (int t = 0; t < 128; t += 2)
      *(unsigned int*)&at[tid * 136 + t] =
          f2bf_pk(s2[tid * 132 + t] * inv, s2[tid * 132 + t + 1] * inv);
  }
  __syncthreads();

  v8f oacc[2];
#pragma unroll
  for (int j = 0; j < 2; ++j) oacc[j] = v8f_zero();
  for (int k0 = 0; k0 < 128; k0 += 32) {
    v16bf a = load_a_frag(at, 136, w * 16, k0);
#pragma unroll
    for (int j = 0; j < 2; ++j) {
      v16bf b = load_b_frag_kmajor(vs, 40, k0, j * 16);  // k=t, n=d
      oacc[j] = wmma_bf16(a, b, oacc[j]);
    }
  }
#pragma unroll
  for (int j = 0; j < 2; ++j) {
    const int n  = h * 32 + j * 16 + (lane & 15);
    const int m0 = w * 16 + (lane >> 4) * 8;
#pragma unroll
    for (int r = 0; r < 8; ++r)
      O[((size_t)(m0 + r) * 128 + nb) * 256 + n] = oacc[j][r];
  }
}

// ---------------------------------------------------------------------------
// Kernel 5: Out[r] = LayerNorm(A[r] + B[r]) * w + b, C=256, one row per block.
// ---------------------------------------------------------------------------
__global__ __launch_bounds__(256) void k_add_ln(const float* __restrict__ A,
                                                const float* __restrict__ Bv,
                                                const float* __restrict__ w,
                                                const float* __restrict__ b,
                                                float* __restrict__ Out) {
  __shared__ float red[256];
  const int r = blockIdx.x, c = threadIdx.x;
  const float v = A[(size_t)r * 256 + c] + Bv[(size_t)r * 256 + c];
  red[c] = v;
  __syncthreads();
  for (int s = 128; s > 0; s >>= 1) {
    if (c < s) red[c] += red[c + s];
    __syncthreads();
  }
  const float mean = red[0] * (1.0f / 256.0f);
  __syncthreads();
  const float dv = v - mean;
  red[c] = dv * dv;
  __syncthreads();
  for (int s = 128; s > 0; s >>= 1) {
    if (c < s) red[c] += red[c + s];
    __syncthreads();
  }
  const float var = red[0] * (1.0f / 256.0f);
  const float inv = rsqrtf(var + 1e-5f);
  Out[(size_t)r * 256 + c] = dv * inv * w[c] + b[c];
}

// ---------------------------------------------------------------------------
// Host orchestration
// ---------------------------------------------------------------------------
static constexpr unsigned LATENT_SMEM =
    (128 * 264 + 64 * 264 + 128 * 72) * 2 + (128 * 68 + 256 + 64) * 4;
static constexpr unsigned MHA_SMEM =
    (3 * 128 * 40 + 128 * 136) * 2 + 128 * 132 * 4;

extern "C" void kernel_launch(void* const* d_in, const int* in_sizes, int n_in,
                              void* d_out, int out_size, void* d_ws, size_t ws_size,
                              hipStream_t stream) {
  (void)in_sizes; (void)n_in; (void)out_size; (void)ws_size;

  const float* x     = (const float*)d_in[0];
  const float* W_lin = (const float*)d_in[1];
  const float* b_lin = (const float*)d_in[2];
  const float* emb   = (const float*)d_in[3];
  const float* inW   = (const float*)d_in[4];
  const float* inB   = (const float*)d_in[5];
  const float* outW  = (const float*)d_in[6];
  const float* outB  = (const float*)d_in[7];
  const float* linW  = (const float*)d_in[8];
  const float* linB  = (const float*)d_in[9];
  const float* ln1w  = (const float*)d_in[10];
  const float* ln1b  = (const float*)d_in[11];
  const float* ln2w  = (const float*)d_in[12];
  const float* ln2b  = (const float*)d_in[13];

  char* ws = (char*)d_ws;
  unsigned short* X4 = (unsigned short*)ws;                  // 64 MiB (bf16)
  float* Y   = (float*)(ws + ((size_t)64  << 20));           // 16 MiB
  float* Y2  = (float*)(ws + ((size_t)80  << 20));           // 16 MiB
  float* QKV = (float*)(ws + ((size_t)96  << 20));           // 48 MiB
  float* O   = (float*)(ws + ((size_t)144 << 20));           // 16 MiB
  float* O2  = (float*)(ws + ((size_t)160 << 20));           // 16 MiB (also T1)

  float* outY    = (float*)d_out;                 // (B, G*K, C) f32
  float* outAttn = (float*)d_out + 4194304;       // (B, G, K, T) f32

  k_proj<<<dim3(1024, 2), 256, 0, stream>>>(x, W_lin, b_lin, X4, 131072, 256, 256);
  k_latent<<<128, 256, LATENT_SMEM, stream>>>(X4, emb, Y, outAttn);

  float* ycur = Y;
  for (int l = 0; l < 2; ++l) {
    float* ymid = (ycur == Y) ? Y2 : Y;
    k_gemm_nt<<<dim3(128, 6), 256, 0, stream>>>(
        ycur, inW + (size_t)l * 768 * 256, inB + (size_t)l * 768, QKV,
        16384, 256, 768);
    k_mha<<<1024, 256, MHA_SMEM, stream>>>(QKV, O);
    k_gemm_nt<<<dim3(128, 2), 256, 0, stream>>>(
        O, outW + (size_t)l * 256 * 256, outB + (size_t)l * 256, O2,
        16384, 256, 256);
    k_add_ln<<<16384, 256, 0, stream>>>(O2, ycur, ln1w + (size_t)l * 256,
                                        ln1b + (size_t)l * 256, ymid);
    k_gemm_nt<<<dim3(128, 2), 256, 0, stream>>>(
        ymid, linW + (size_t)l * 256 * 256, linB + (size_t)l * 256, O2,
        16384, 256, 256);
    float* ynext = (l == 1) ? outY : ycur;
    k_add_ln<<<16384, 256, 0, stream>>>(O2, ymid, ln2w + (size_t)l * 256,
                                        ln2b + (size_t)l * 256, ynext);
    ycur = ynext;
  }
}
